// Transformer_13305808683566
// MI455X (gfx1250) — matrix operand order repaired and checked by reading, not compiled, not run on hardware
//
#include <hip/hip_runtime.h>
#include <math.h>

// ---------------------------------------------------------------------------
// Types for CDNA5 WMMA (wave32): v_wmma_f32_16x16x32_bf16
// ---------------------------------------------------------------------------
typedef __bf16 bf16;
typedef __attribute__((ext_vector_type(16))) __bf16 v16bf;
typedef __attribute__((ext_vector_type(8)))  __bf16 v8bf;
typedef __attribute__((ext_vector_type(8)))  float  v8f;

__device__ __forceinline__ bf16 f2bf(float f) {
    union { float f; unsigned u; } x; x.f = f;
    unsigned r = (x.u + 0x7FFFu + ((x.u >> 16) & 1u)) >> 16;  // RNE
    unsigned short s = (unsigned short)r;
    return __builtin_bit_cast(bf16, s);
}
__device__ __forceinline__ float bf2f(bf16 b) {
    unsigned short s = __builtin_bit_cast(unsigned short, b);
    union { unsigned u; float f; } x; x.u = ((unsigned)s) << 16;
    return x.f;
}

// ---------------------------------------------------------------------------
// Model constants
// ---------------------------------------------------------------------------
#define CB   8
#define CT   256
#define CC   384
#define CH   6
#define CD   64
#define CL   6
#define CV   32000
#define CM   2048              // B*T rows
#define NLOGITS ((size_t)CM * CV)

// ---------------------------------------------------------------------------
// Register-blocked WMMA GEMM:
//   out[M,N] = A[M,K](bf16,row-major) x Bt[N,K](bf16,row-major = B^T)
// with fused bias / residual / relu epilogue.
//
// Block = 256 threads = 8 waves arranged 4(M) x 2(N); block tile 128x128.
// Each wave computes a 32x64 strip: 2 A-frags x 4 B-frags -> 8 WMMAs/k-step,
// ~21 FLOP/byte of cache traffic (vs 8 for the 1-tile version).
//
// A-operand layout (16-bit A 16x32): lane l: row = l&15, half = l>>4,
//   elems 0..15 = K (k0 + half*16 .. +15)  (two 16B loads).
// B-operand layout (32x16): lane l: col = l&15, half = l>>4,
//   elems 0..15 = K (k0 + half*16 .. +15)  -> one contiguous 32B load of Bt.
// D layout (16x16 f32): lane l: N = l&15, VGPR r -> M = (l>>4)*8 + r.
// ---------------------------------------------------------------------------
__global__ void __launch_bounds__(256)
gemm_bf16_wmma(const bf16* __restrict__ A, const bf16* __restrict__ Bt,
               const float* __restrict__ bias, const float* __restrict__ resid,
               float* __restrict__ outF, bf16* __restrict__ outB,
               int M, int N, int K, int relu)
{
    const int lane  = threadIdx.x & 31;
    const int wave  = threadIdx.x >> 5;
    const int l16   = lane & 15;
    const int half  = lane >> 4;
    const int tileM = blockIdx.y * 128 + (wave & 3) * 32;   // wave: 32 rows
    const int tileN = blockIdx.x * 128 + (wave >> 2) * 64;  // wave: 64 cols

    const bf16* arow0 = A  + (size_t)(tileM +      l16) * K;
    const bf16* arow1 = A  + (size_t)(tileM + 16 + l16) * K;
    const bf16* brow0 = Bt + (size_t)(tileN +      l16) * K;
    const bf16* brow1 = Bt + (size_t)(tileN + 16 + l16) * K;
    const bf16* brow2 = Bt + (size_t)(tileN + 32 + l16) * K;
    const bf16* brow3 = Bt + (size_t)(tileN + 48 + l16) * K;

    v8f acc[2][4];
    #pragma unroll
    for (int mi = 0; mi < 2; ++mi)
        #pragma unroll
        for (int ni = 0; ni < 4; ++ni)
            acc[mi][ni] = (v8f){};

    const int ka = half * 16;   // A chunk offset within k-step
    const int kb = half * 16;   // B chunk offset within k-step
    for (int k0 = 0; k0 < K; k0 += 32) {
        union { v16bf v; v8bf h[2]; } a0, a1;
        a0.h[0] = *(const v8bf*)(arow0 + k0 + ka);
        a0.h[1] = *(const v8bf*)(arow0 + k0 + 8 + ka);
        a1.h[0] = *(const v8bf*)(arow1 + k0 + ka);
        a1.h[1] = *(const v8bf*)(arow1 + k0 + 8 + ka);
        v16bf b0 = *(const v16bf*)(brow0 + k0 + kb);
        v16bf b1 = *(const v16bf*)(brow1 + k0 + kb);
        v16bf b2 = *(const v16bf*)(brow2 + k0 + kb);
        v16bf b3 = *(const v16bf*)(brow3 + k0 + kb);

        acc[0][0] = __builtin_amdgcn_wmma_f32_16x16x32_bf16(false, a0.v, false, b0, (short)0, acc[0][0], false, false);
        acc[0][1] = __builtin_amdgcn_wmma_f32_16x16x32_bf16(false, a0.v, false, b1, (short)0, acc[0][1], false, false);
        acc[0][2] = __builtin_amdgcn_wmma_f32_16x16x32_bf16(false, a0.v, false, b2, (short)0, acc[0][2], false, false);
        acc[0][3] = __builtin_amdgcn_wmma_f32_16x16x32_bf16(false, a0.v, false, b3, (short)0, acc[0][3], false, false);
        acc[1][0] = __builtin_amdgcn_wmma_f32_16x16x32_bf16(false, a1.v, false, b0, (short)0, acc[1][0], false, false);
        acc[1][1] = __builtin_amdgcn_wmma_f32_16x16x32_bf16(false, a1.v, false, b1, (short)0, acc[1][1], false, false);
        acc[1][2] = __builtin_amdgcn_wmma_f32_16x16x32_bf16(false, a1.v, false, b2, (short)0, acc[1][2], false, false);
        acc[1][3] = __builtin_amdgcn_wmma_f32_16x16x32_bf16(false, a1.v, false, b3, (short)0, acc[1][3], false, false);
    }

    #pragma unroll
    for (int ni = 0; ni < 4; ++ni) {
        const int ncol = tileN + ni * 16 + l16;
        const float bval = bias ? bias[ncol] : 0.0f;
        #pragma unroll
        for (int mi = 0; mi < 2; ++mi) {
            #pragma unroll
            for (int r = 0; r < 8; ++r) {
                const int m = tileM + mi * 16 + half * 8 + r;
                const size_t o = (size_t)m * N + ncol;
                float v = acc[mi][ni][r] + bval;
                if (resid) v += resid[o];
                if (relu)  v = fmaxf(v, 0.0f);
                if (outF)  outF[o] = v;
                if (outB)  outB[o] = f2bf(v);
            }
        }
    }
}

// ---------------------------------------------------------------------------
// Weight conversion kernels (fp32 -> bf16, optionally transposed to [N,K])
// ---------------------------------------------------------------------------
__global__ void __launch_bounds__(256)
convert_bf16_kernel(const float* __restrict__ in, bf16* __restrict__ out, int n)
{
    int i = blockIdx.x * 256 + threadIdx.x;
    if (i < n) out[i] = f2bf(in[i]);
}

__global__ void __launch_bounds__(256)
transpose_bf16_kernel(const float* __restrict__ in, bf16* __restrict__ out,
                      int K, int N)   // in: [K,N] row-major -> out: [N,K]
{
    int i = blockIdx.x * 256 + threadIdx.x;
    if (i < K * N) {
        int nn = i / K, kk = i - nn * K;
        out[i] = f2bf(in[(size_t)kk * N + nn]);
    }
}

// ---------------------------------------------------------------------------
// Embedding: x[row, c] = tok_emb[idx[row], c] + pos_emb[row%T, c]
// ---------------------------------------------------------------------------
__global__ void __launch_bounds__(128)
embed_kernel(const int* __restrict__ idx, const float* __restrict__ tok,
             const float* __restrict__ pos, float* __restrict__ x)
{
    const int row = blockIdx.x;
    const int t   = row & (CT - 1);
    const int tid = idx[row];
    for (int c = threadIdx.x; c < CC; c += 128)
        x[(size_t)row * CC + c] = tok[(size_t)tid * CC + c] + pos[(size_t)t * CC + c];
}

// ---------------------------------------------------------------------------
// LayerNorm (fp32 in) -> bf16 out, one row (C=384) per block of 128 threads
// ---------------------------------------------------------------------------
__global__ void __launch_bounds__(128)
layernorm_to_bf16(const float* __restrict__ x, const float* __restrict__ w,
                  const float* __restrict__ b, bf16* __restrict__ out)
{
    __shared__ float rs[4], rs2[4];
    const int row = blockIdx.x;
    const float* xr = x + (size_t)row * CC;
    const int t = threadIdx.x;
    float v0 = xr[t], v1 = xr[t + 128], v2 = xr[t + 256];
    float s  = v0 + v1 + v2;
    float s2 = v0 * v0 + v1 * v1 + v2 * v2;
    for (int o = 16; o > 0; o >>= 1) { s += __shfl_xor(s, o, 32); s2 += __shfl_xor(s2, o, 32); }
    if ((t & 31) == 0) { rs[t >> 5] = s; rs2[t >> 5] = s2; }
    __syncthreads();
    s  = rs[0] + rs[1] + rs[2] + rs[3];
    s2 = rs2[0] + rs2[1] + rs2[2] + rs2[3];
    const float mu  = s * (1.0f / CC);
    const float var = s2 * (1.0f / CC) - mu * mu;
    const float inv = rsqrtf(var + 1e-5f);
    bf16* orow = out + (size_t)row * CC;
    orow[t]       = f2bf((v0 - mu) * inv * w[t]       + b[t]);
    orow[t + 128] = f2bf((v1 - mu) * inv * w[t + 128] + b[t + 128]);
    orow[t + 256] = f2bf((v2 - mu) * inv * w[t + 256] + b[t + 256]);
}

// ---------------------------------------------------------------------------
// Causal attention, one (b,h) per block. K staged in LDS as bf16 (32 KB).
// q,k,v,o are [2048, 384] bf16 (head h occupies cols h*64..h*64+63).
// ---------------------------------------------------------------------------
__global__ void __launch_bounds__(256)
attention_kernel(const bf16* __restrict__ q, const bf16* __restrict__ k,
                 const bf16* __restrict__ v, bf16* __restrict__ o)
{
    __shared__ bf16  ks[CT * CD];     // 32 KB
    __shared__ float qs[CD];
    __shared__ float as[CT];
    __shared__ float red[8];
    __shared__ float op[4 * CD];

    const int b = blockIdx.x / CH;
    const int h = blockIdx.x % CH;
    const int tid  = threadIdx.x;
    const int lane = tid & 31, wid = tid >> 5;
    const size_t base = ((size_t)b * CT) * CC + (size_t)h * CD;

    for (int i = tid; i < CT * CD; i += 256) {
        int j = i >> 6, d = i & 63;
        ks[i] = k[base + (size_t)j * CC + d];
    }
    __syncthreads();

    for (int t = 0; t < CT; ++t) {
        if (tid < CD) qs[tid] = bf2f(q[base + (size_t)t * CC + tid]);
        __syncthreads();

        float s = -3.0e38f;
        if (tid <= t) {
            float a = 0.0f;
            const bf16* kr = ks + tid * CD;
            #pragma unroll 8
            for (int d = 0; d < CD; ++d) a += qs[d] * bf2f(kr[d]);
            s = a * 0.125f;                       // 1/sqrt(64)
        }
        // block max
        float m = s;
        for (int off = 16; off > 0; off >>= 1) m = fmaxf(m, __shfl_xor(m, off, 32));
        if (lane == 0) red[wid] = m;
        __syncthreads();
        m = red[0];
        #pragma unroll
        for (int i = 1; i < 8; ++i) m = fmaxf(m, red[i]);

        float p = (tid <= t) ? __expf(s - m) : 0.0f;
        as[tid] = p;
        float su = p;
        for (int off = 16; off > 0; off >>= 1) su += __shfl_xor(su, off, 32);
        __syncthreads();                           // red reads + as[] writes done
        if (lane == 0) red[wid] = su;
        __syncthreads();
        su = red[0] + red[1] + red[2] + red[3] + red[4] + red[5] + red[6] + red[7];
        const float inv = 1.0f / su;

        // out[d] = sum_j a_j * v[j][d], split j over 4 groups
        const int d = tid & 63, g = tid >> 6;
        float acc = 0.0f;
        for (int j = g; j <= t; j += 4)
            acc += as[j] * bf2f(v[base + (size_t)j * CC + d]);
        op[g * CD + d] = acc;
        __syncthreads();
        if (tid < CD) {
            float tot = (op[tid] + op[CD + tid] + op[2 * CD + tid] + op[3 * CD + tid]) * inv;
            o[base + (size_t)t * CC + tid] = f2bf(tot);
        }
        __syncthreads();
    }
}

// ---------------------------------------------------------------------------
// Loss: per-row log-softmax NLL, then mean. Deterministic (no atomics).
// ---------------------------------------------------------------------------
__global__ void __launch_bounds__(256)
nll_row_kernel(const float* __restrict__ logits, const int* __restrict__ targets,
               float* __restrict__ nll)
{
    __shared__ float red[8];
    const int row = blockIdx.x;
    const float* lr = logits + (size_t)row * CV;
    const int tid = threadIdx.x, lane = tid & 31, wid = tid >> 5;

    float mx = -3.0e38f;
    for (int i = tid; i < CV; i += 256) mx = fmaxf(mx, lr[i]);
    for (int o = 16; o > 0; o >>= 1) mx = fmaxf(mx, __shfl_xor(mx, o, 32));
    if (lane == 0) red[wid] = mx;
    __syncthreads();
    mx = red[0];
    #pragma unroll
    for (int i = 1; i < 8; ++i) mx = fmaxf(mx, red[i]);

    float su = 0.0f;
    for (int i = tid; i < CV; i += 256) su += __expf(lr[i] - mx);
    for (int o = 16; o > 0; o >>= 1) su += __shfl_xor(su, o, 32);
    __syncthreads();
    if (lane == 0) red[wid] = su;
    __syncthreads();
    if (tid == 0) {
        float tot = 0.0f;
        #pragma unroll
        for (int i = 0; i < 8; ++i) tot += red[i];
        nll[row] = (mx + __logf(tot)) - lr[targets[row]];
    }
}

__global__ void __launch_bounds__(256)
loss_mean_kernel(const float* __restrict__ nll, float* __restrict__ out)
{
    __shared__ float red[8];
    const int tid = threadIdx.x;
    float s = 0.0f;
    for (int i = tid; i < CM; i += 256) s += nll[i];
    for (int o = 16; o > 0; o >>= 1) s += __shfl_xor(s, o, 32);
    if ((tid & 31) == 0) red[tid >> 5] = s;
    __syncthreads();
    if (tid == 0) {
        float t = 0.0f;
        #pragma unroll
        for (int i = 0; i < 8; ++i) t += red[i];
        out[0] = t / (float)CM;
    }
}

// ---------------------------------------------------------------------------
// Workspace layout (bytes, all offsets 256-aligned)
// ---------------------------------------------------------------------------
static const size_t OFF_X   = 0;                                       // f32 [2048,384]
static const size_t OFF_H   = OFF_X  + (size_t)CM * CC * 4;            // bf16 [2048,384]
static const size_t OFF_Q   = OFF_H  + (size_t)CM * CC * 2;
static const size_t OFF_K   = OFF_Q  + (size_t)CM * CC * 2;
static const size_t OFF_V   = OFF_K  + (size_t)CM * CC * 2;
static const size_t OFF_O   = OFF_V  + (size_t)CM * CC * 2;
static const size_t OFF_U   = OFF_O  + (size_t)CM * CC * 2;            // bf16 [2048,1536]
static const size_t OFF_WQ  = OFF_U  + (size_t)CM * 4 * CC * 2;
static const size_t OFF_WK  = OFF_WQ + (size_t)CL * CC * CC * 2;
static const size_t OFF_WV  = OFF_WK + (size_t)CL * CC * CC * 2;
static const size_t OFF_WO  = OFF_WV + (size_t)CL * CC * CC * 2;
static const size_t OFF_W1  = OFF_WO + (size_t)CL * CC * CC * 2;
static const size_t OFF_W2  = OFF_W1 + (size_t)CL * CC * 4 * CC * 2;
static const size_t OFF_LM  = OFF_W2 + (size_t)CL * 4 * CC * CC * 2;
static const size_t OFF_NLL = OFF_LM + (size_t)CV * CC * 2;

extern "C" void kernel_launch(void* const* d_in, const int* in_sizes, int n_in,
                              void* d_out, int out_size, void* d_ws, size_t ws_size,
                              hipStream_t stream)
{
    (void)in_sizes; (void)n_in; (void)out_size; (void)ws_size;

    const int*   idx     = (const int*)  d_in[0];
    const int*   targets = (const int*)  d_in[1];
    const float* tok_emb = (const float*)d_in[2];
    const float* pos_emb = (const float*)d_in[3];
    const float* ln1_w   = (const float*)d_in[4];
    const float* ln1_b   = (const float*)d_in[5];
    const float* wq      = (const float*)d_in[6];
    const float* wk      = (const float*)d_in[7];
    const float* wv      = (const float*)d_in[8];
    const float* wo      = (const float*)d_in[9];
    const float* bo      = (const float*)d_in[10];
    const float* ln2_w   = (const float*)d_in[11];
    const float* ln2_b   = (const float*)d_in[12];
    const float* w1      = (const float*)d_in[13];
    const float* b1      = (const float*)d_in[14];
    const float* w2      = (const float*)d_in[15];
    const float* b2      = (const float*)d_in[16];
    const float* lnf_w   = (const float*)d_in[17];
    const float* lnf_b   = (const float*)d_in[18];
    const float* lm_w    = (const float*)d_in[19];
    const float* lm_b    = (const float*)d_in[20];

    char* ws = (char*)d_ws;
    float* x    = (float*)(ws + OFF_X);
    bf16*  hbf  = (bf16*)(ws + OFF_H);
    bf16*  qbf  = (bf16*)(ws + OFF_Q);
    bf16*  kbf  = (bf16*)(ws + OFF_K);
    bf16*  vbf  = (bf16*)(ws + OFF_V);
    bf16*  obf  = (bf16*)(ws + OFF_O);
    bf16*  ubf  = (bf16*)(ws + OFF_U);
    bf16*  wqT  = (bf16*)(ws + OFF_WQ);
    bf16*  wkT  = (bf16*)(ws + OFF_WK);
    bf16*  wvT  = (bf16*)(ws + OFF_WV);
    bf16*  woT  = (bf16*)(ws + OFF_WO);
    bf16*  w1T  = (bf16*)(ws + OFF_W1);
    bf16*  w2T  = (bf16*)(ws + OFF_W2);
    bf16*  lmB  = (bf16*)(ws + OFF_LM);
    float* nll  = (float*)(ws + OFF_NLL);
    float* logits = (float*)d_out;
    float* lossp  = (float*)d_out + NLOGITS;

    const int nCC  = CC * CC;                 // 147456
    const int nCC4 = CC * 4 * CC;             // 589824

    // --- weight conversion (fp32 -> bf16, transposed to [N,K]) -------------
    for (int l = 0; l < CL; ++l) {
        const size_t oq = (size_t)l * nCC;
        transpose_bf16_kernel<<<(nCC + 255) / 256, 256, 0, stream>>>(wq + oq, wqT + oq, CC, CC);
        transpose_bf16_kernel<<<(nCC + 255) / 256, 256, 0, stream>>>(wk + oq, wkT + oq, CC, CC);
        transpose_bf16_kernel<<<(nCC + 255) / 256, 256, 0, stream>>>(wv + oq, wvT + oq, CC, CC);
        transpose_bf16_kernel<<<(nCC + 255) / 256, 256, 0, stream>>>(wo + oq, woT + oq, CC, CC);
        const size_t om = (size_t)l * nCC4;
        transpose_bf16_kernel<<<(nCC4 + 255) / 256, 256, 0, stream>>>(w1 + om, w1T + om, CC, 4 * CC);
        transpose_bf16_kernel<<<(nCC4 + 255) / 256, 256, 0, stream>>>(w2 + om, w2T + om, 4 * CC, CC);
    }
    convert_bf16_kernel<<<((CV * CC) + 255) / 256, 256, 0, stream>>>(lm_w, lmB, CV * CC);

    // --- embedding ---------------------------------------------------------
    embed_kernel<<<CM, 128, 0, stream>>>(idx, tok_emb, pos_emb, x);

    const dim3 g384(CC / 128, CM / 128);        // (3, 16)
    const dim3 g1536(4 * CC / 128, CM / 128);   // (12, 16)

    // --- transformer layers ------------------------------------------------
    for (int l = 0; l < CL; ++l) {
        layernorm_to_bf16<<<CM, 128, 0, stream>>>(x, ln1_w + l * CC, ln1_b + l * CC, hbf);

        gemm_bf16_wmma<<<g384, 256, 0, stream>>>(hbf, wqT + (size_t)l * nCC,
            nullptr, nullptr, nullptr, qbf, CM, CC, CC, 0);
        gemm_bf16_wmma<<<g384, 256, 0, stream>>>(hbf, wkT + (size_t)l * nCC,
            nullptr, nullptr, nullptr, kbf, CM, CC, CC, 0);
        gemm_bf16_wmma<<<g384, 256, 0, stream>>>(hbf, wvT + (size_t)l * nCC,
            nullptr, nullptr, nullptr, vbf, CM, CC, CC, 0);

        attention_kernel<<<CB * CH, 256, 0, stream>>>(qbf, kbf, vbf, obf);

        // x = x + o @ wo + bo
        gemm_bf16_wmma<<<g384, 256, 0, stream>>>(obf, woT + (size_t)l * nCC,
            bo + l * CC, x, x, nullptr, CM, CC, CC, 0);

        layernorm_to_bf16<<<CM, 128, 0, stream>>>(x, ln2_w + l * CC, ln2_b + l * CC, hbf);

        // u = relu(h2 @ w1 + b1)
        gemm_bf16_wmma<<<g1536, 256, 0, stream>>>(hbf, w1T + (size_t)l * nCC4,
            b1 + l * 4 * CC, nullptr, nullptr, ubf, CM, 4 * CC, CC, 1);
        // x = x + u @ w2 + b2
        gemm_bf16_wmma<<<g384, 256, 0, stream>>>(ubf, w2T + (size_t)l * nCC4,
            b2 + l * CC, x, x, nullptr, CM, CC, 4 * CC, 0);
    }

    // --- final LN + LM head ------------------------------------------------
    layernorm_to_bf16<<<CM, 128, 0, stream>>>(x, lnf_w, lnf_b, hbf);
    const dim3 gLM(CV / 128, CM / 128);         // (250, 16)
    gemm_bf16_wmma<<<gLM, 256, 0, stream>>>(hbf, lmB, lm_b,
        nullptr, logits, nullptr, CM, CV, CC, 0);

    // --- loss --------------------------------------------------------------
    nll_row_kernel<<<CM, 256, 0, stream>>>(logits, targets, nll);
    loss_mean_kernel<<<1, 256, 0, stream>>>(nll, lossp);
}
